// DAMM_CKAM_57337813401752
// MI455X (gfx1250) — matrix operand order repaired and checked by reading, not compiled, not run on hardware
//
#include <hip/hip_runtime.h>
#include <hip/hip_bf16.h>

typedef __bf16 bf16_t;
typedef __attribute__((ext_vector_type(16))) __bf16 v16bf;
typedef __attribute__((ext_vector_type(8)))  __bf16 v8bf;
typedef __attribute__((ext_vector_type(8)))  float  v8f;

#define N_SP 4096   // H*W
#define C_CH 512
#define D_Q  64
#define OUTC 256
#define NB   4

// ---------------------------------------------------------------------------
// WMMA helpers (CDNA5 16x16x32 bf16, f32 accumulate)
// ---------------------------------------------------------------------------
__device__ inline v8f wmma_bf16(v16bf a, v16bf b, v8f c) {
  return __builtin_amdgcn_wmma_f32_16x16x32_bf16(
      /*neg_a=*/false, a, /*neg_b=*/false, b,
      /*c_mod=*/(short)0, c, /*reuse_a=*/false, /*reuse_b=*/false);
}

// A fragment: 16(M) x 32(K), source row-major, row stride ld (elems, mult of 8).
// Lane: row m = lane%16; K = {8h..8h+7} U {16+8h..16+8h+7}, h = lane/16.
// Two explicit 16-byte loads per lane.
__device__ inline v16bf frag_a(const bf16_t* p, int ld, int lane) {
  const int l16 = lane & 15, h = (lane >> 4) & 1;
  const bf16_t* r = p + (size_t)l16 * ld + h * 8;
  v8bf lo = *(const v8bf*)(r);
  v8bf hi = *(const v8bf*)(r + 16);
  v16bf a;
#pragma unroll
  for (int i = 0; i < 8; ++i) { a[i] = lo[i]; a[8 + i] = hi[i]; }
  return a;
}

// B fragment: 32(K) x 16(N), source stored TRANSPOSED as [N][K] row-major,
// row stride ld (mult of 8). Lane: col n = lane%16; K = {8h..8h+7} U {16+8h..16+8h+7}.
__device__ inline v16bf frag_b(const bf16_t* pT, int ld, int lane) {
  const int l16 = lane & 15, h = (lane >> 4) & 1;
  const bf16_t* r = pT + (size_t)l16 * ld + h * 8;
  v8bf lo = *(const v8bf*)(r);
  v8bf hi = *(const v8bf*)(r + 16);
  v16bf b;
#pragma unroll
  for (int i = 0; i < 8; ++i) { b[i] = lo[i]; b[8 + i] = hi[i]; }
  return b;
}

__device__ inline v8f v8f_zero() {
  v8f z;
#pragma unroll
  for (int i = 0; i < 8; ++i) z[i] = 0.0f;
  return z;
}

// ---------------------------------------------------------------------------
// K0a: cast fp32 -> bf16 (conv1_w)
// ---------------------------------------------------------------------------
__global__ __launch_bounds__(256) void cast_f32_bf16_kernel(
    const float* __restrict__ src, bf16_t* __restrict__ dst, int n) {
  int i = blockIdx.x * 256 + threadIdx.x;
  if (i < n) dst[i] = (bf16_t)src[i];
}

// ---------------------------------------------------------------------------
// K0b: x -> xb [B][C][N] bf16  AND  xbT [B][N][C] bf16 (LDS-tiled transpose).
// block (32,8), grid (N/32, C/32, B)
// ---------------------------------------------------------------------------
__global__ __launch_bounds__(256) void cast_transpose_kernel(
    const float* __restrict__ x, bf16_t* __restrict__ xb,
    bf16_t* __restrict__ xbT) {
  const int b = blockIdx.z;
  const int n0 = blockIdx.x * 32, c0 = blockIdx.y * 32;
  const int tx = threadIdx.x, ty = threadIdx.y;
  __shared__ float tile[32][33];
#pragma unroll
  for (int t = 0; t < 4; ++t) {
    int cl = ty + t * 8;
    size_t idx = ((size_t)b * C_CH + c0 + cl) * N_SP + n0 + tx;
    float v = x[idx];
    xb[idx] = (bf16_t)v;
    tile[cl][tx] = v;
  }
  __syncthreads();
#pragma unroll
  for (int t = 0; t < 4; ++t) {
    int nl = ty + t * 8;
    xbT[((size_t)b * N_SP + n0 + nl) * C_CH + c0 + tx] = (bf16_t)tile[tx][nl];
  }
}

// ---------------------------------------------------------------------------
// K0c: stack q_w/k_w/v_w into Wqkv[640][512] bf16 + bias640 f32
// ---------------------------------------------------------------------------
__global__ __launch_bounds__(256) void pack_w_kernel(
    const float* __restrict__ qw, const float* __restrict__ kw,
    const float* __restrict__ vw, const float* __restrict__ qbias,
    const float* __restrict__ kbias, const float* __restrict__ vbias,
    bf16_t* __restrict__ wqkv, float* __restrict__ bias640) {
  int i = blockIdx.x * 256 + threadIdx.x;
  if (i < 640 * 512) {
    int o = i >> 9, c = i & 511;
    float v;
    if (o < 64)        v = qw[o * 512 + c];
    else if (o < 128)  v = kw[(o - 64) * 512 + c];
    else               v = vw[(o - 128) * 512 + c];
    wqkv[i] = (bf16_t)v;
  }
  if (i < 640) {
    float bv = (i < 64) ? qbias[i] : (i < 128 ? kbias[i - 64] : vbias[i - 128]);
    bias640[i] = bv;
  }
}

// ---------------------------------------------------------------------------
// K1: QKV projection, LDS-free. A = Wqkv[o][c] (global), B = xT[p][c] (global).
// M=640, N=4096, K=512 per batch; tile 64x128, 8 waves.
// q/k written TRANSPOSED as qkT[b][sel][p][d] (packed 16B stores);
// v written as vB[b][c][n].
// ---------------------------------------------------------------------------
__global__ __launch_bounds__(256) void qkv_kernel(
    const bf16_t* __restrict__ wqkv, const bf16_t* __restrict__ xbT,
    const float* __restrict__ bias640, bf16_t* __restrict__ qkT,
    bf16_t* __restrict__ vB) {
  const int b = blockIdx.z;
  const int m0 = blockIdx.y * 64;
  const int n0 = blockIdx.x * 128;
  const int tid = threadIdx.x, lane = tid & 31, wave = tid >> 5;
  const int l16 = lane & 15, h = (lane >> 4) & 1;
  const int mi = wave & 3, nh = wave >> 2;

  const bf16_t* xT = xbT + (size_t)b * N_SP * C_CH;

  v8f acc[4];
#pragma unroll
  for (int j = 0; j < 4; ++j) acc[j] = v8f_zero();

  for (int k0 = 0; k0 < C_CH; k0 += 32) {
    v16bf a = frag_a(wqkv + (size_t)(m0 + mi * 16) * C_CH + k0, C_CH, lane);
#pragma unroll
    for (int j = 0; j < 4; ++j) {
      v16bf bf = frag_b(xT + (size_t)(n0 + nh * 64 + j * 16) * C_CH + k0,
                        C_CH, lane);
      acc[j] = wmma_bf16(a, bf, acc[j]);
    }
  }

  if (m0 < 128) {
    // q (sel=0) or k (sel=1): transposed [p][d] layout, 16B packed stores
    const int sel = m0 >> 6;
#pragma unroll
    for (int j = 0; j < 4; ++j) {
      int p = n0 + nh * 64 + j * 16 + l16;
      v8bf pk;
#pragma unroll
      for (int r = 0; r < 8; ++r) {
        int o = m0 + mi * 16 + r + 8 * h;
        pk[r] = (bf16_t)(acc[j][r] + bias640[o]);
      }
      *(v8bf*)(qkT + ((size_t)(b * 2 + sel) * N_SP + p) * D_Q +
               (mi * 16 + 8 * h)) = pk;
    }
  } else {
#pragma unroll
    for (int j = 0; j < 4; ++j)
#pragma unroll
      for (int r = 0; r < 8; ++r) {
        int o = m0 + mi * 16 + r + 8 * h;
        int p = n0 + nh * 64 + j * 16 + l16;
        vB[((size_t)b * C_CH + (o - 128)) * N_SP + p] =
            (bf16_t)(acc[j][r] + bias640[o]);
      }
  }
}

// ---------------------------------------------------------------------------
// K2: fused PAM flash attention. Q/K fragments direct from global qkT[p][64];
// V fragments direct from global vB[c][n]. Only P goes through LDS (2 KB).
// 8 waves, each owns a 64-wide slice of the 512 value channels; S computed
// redundantly per wave -> softmax needs no cross-wave traffic.
// Writes x1 = gamma_pam * out + x into s12 (float4 stores).
// ---------------------------------------------------------------------------
__global__ __launch_bounds__(256) void pam_kernel(
    const bf16_t* __restrict__ qkT, const bf16_t* __restrict__ vB,
    const float* __restrict__ x, const float* __restrict__ gamma_pam,
    float* __restrict__ s12) {
  const int b = blockIdx.y;
  const int m0 = blockIdx.x * 32;
  const int tid = threadIdx.x, lane = tid & 31, wave = tid >> 5;
  const int l16 = lane & 15, h = (lane >> 4) & 1;
  const int c0 = wave * 64;

  const bf16_t* qT = qkT + (size_t)(b * 2 + 0) * N_SP * D_Q;  // [m][d]
  const bf16_t* kT = qkT + (size_t)(b * 2 + 1) * N_SP * D_Q;  // [n][d]
  const bf16_t* vb = vB + (size_t)b * C_CH * N_SP;            // [c][n]

  __shared__ bf16_t Pl[32 * 32];  // [m][n] probabilities (bf16)

  v16bf qa[2][2];
#pragma unroll
  for (int mi = 0; mi < 2; ++mi)
#pragma unroll
    for (int kd = 0; kd < 2; ++kd)
      qa[mi][kd] =
          frag_a(qT + (size_t)(m0 + mi * 16) * D_Q + kd * 32, D_Q, lane);

  float mrun[2][8], lrun[2][8];
#pragma unroll
  for (int mi = 0; mi < 2; ++mi)
#pragma unroll
    for (int r = 0; r < 8; ++r) { mrun[mi][r] = -3.0e38f; lrun[mi][r] = 0.0f; }

  v8f oacc[2][4];
#pragma unroll
  for (int mi = 0; mi < 2; ++mi)
#pragma unroll
    for (int cj = 0; cj < 4; ++cj) oacc[mi][cj] = v8f_zero();

  for (int n0 = 0; n0 < N_SP; n0 += 32) {
    // S = Q K^T  (identical across waves; fragments direct from global)
    v8f s[2][2];
#pragma unroll
    for (int mi = 0; mi < 2; ++mi)
#pragma unroll
      for (int ni = 0; ni < 2; ++ni) {
        v8f a = v8f_zero();
#pragma unroll
        for (int kd = 0; kd < 2; ++kd) {
          v16bf bf = frag_b(kT + (size_t)(n0 + ni * 16) * D_Q + kd * 32,
                            D_Q, lane);
          a = wmma_bf16(qa[mi][kd], bf, a);
        }
        s[mi][ni] = a;
      }

    // Online softmax (row stats via 16-lane xor reductions, per half)
#pragma unroll
    for (int mi = 0; mi < 2; ++mi) {
      float cfac[8];
#pragma unroll
      for (int r = 0; r < 8; ++r) {
        float v = fmaxf(s[mi][0][r], s[mi][1][r]);
#pragma unroll
        for (int msk = 1; msk < 16; msk <<= 1)
          v = fmaxf(v, __shfl_xor(v, msk, 32));
        float nm = fmaxf(mrun[mi][r], v);
        cfac[r] = __expf(mrun[mi][r] - nm);
        mrun[mi][r] = nm;
        float p0 = __expf(s[mi][0][r] - nm);
        float p1 = __expf(s[mi][1][r] - nm);
        s[mi][0][r] = p0;
        s[mi][1][r] = p1;
        float rs = p0 + p1;
#pragma unroll
        for (int msk = 1; msk < 16; msk <<= 1)
          rs += __shfl_xor(rs, msk, 32);
        lrun[mi][r] = lrun[mi][r] * cfac[r] + rs;
      }
#pragma unroll
      for (int cj = 0; cj < 4; ++cj)
#pragma unroll
        for (int r = 0; r < 8; ++r) oacc[mi][cj][r] *= cfac[r];
    }

    // Share P through LDS (all waves write identical values: benign race)
    __syncthreads();  // previous iteration's P reads complete
#pragma unroll
    for (int mi = 0; mi < 2; ++mi)
#pragma unroll
      for (int ni = 0; ni < 2; ++ni)
#pragma unroll
        for (int r = 0; r < 8; ++r)
          Pl[(mi * 16 + r + 8 * h) * 32 + ni * 16 + l16] = (bf16_t)s[mi][ni][r];
    __syncthreads();

    // O += P * V  (V B-fragments direct from global: v[c][n], n contiguous)
#pragma unroll
    for (int mi = 0; mi < 2; ++mi) {
      v16bf pa = frag_a(Pl + mi * 16 * 32, 32, lane);
#pragma unroll
      for (int cj = 0; cj < 4; ++cj) {
        v16bf vf = frag_b(vb + (size_t)(c0 + cj * 16) * N_SP + n0, N_SP, lane);
        oacc[mi][cj] = wmma_bf16(pa, vf, oacc[mi][cj]);
      }
    }
  }

  const float g = gamma_pam[0];
#pragma unroll
  for (int mi = 0; mi < 2; ++mi)
#pragma unroll
    for (int cj = 0; cj < 4; ++cj) {
      int c = c0 + cj * 16 + l16;
      size_t base = ((size_t)b * C_CH + c) * N_SP + m0 + mi * 16 + 8 * h;
      float4 x0 = *(const float4*)(x + base);
      float4 x1 = *(const float4*)(x + base + 4);
      float4 o0, o1;
      o0.x = g * (oacc[mi][cj][0] / lrun[mi][0]) + x0.x;
      o0.y = g * (oacc[mi][cj][1] / lrun[mi][1]) + x0.y;
      o0.z = g * (oacc[mi][cj][2] / lrun[mi][2]) + x0.z;
      o0.w = g * (oacc[mi][cj][3] / lrun[mi][3]) + x0.w;
      o1.x = g * (oacc[mi][cj][4] / lrun[mi][4]) + x1.x;
      o1.y = g * (oacc[mi][cj][5] / lrun[mi][5]) + x1.y;
      o1.z = g * (oacc[mi][cj][6] / lrun[mi][6]) + x1.z;
      o1.w = g * (oacc[mi][cj][7] / lrun[mi][7]) + x1.w;
      *(float4*)(s12 + base) = o0;
      *(float4*)(s12 + base + 4) = o1;
    }
}

// ---------------------------------------------------------------------------
// K3: CAM Gram matrix E[c][d] = sum_n x[c][n] x[d][n]. M=N=512, K=4096.
// Both fragments straight from global xb[c][n] (n contiguous). LDS-free.
// ---------------------------------------------------------------------------
__global__ __launch_bounds__(256) void cam_energy_kernel(
    const bf16_t* __restrict__ xb, float* __restrict__ camE) {
  const int b = blockIdx.z;
  const int m0 = blockIdx.y * 64;
  const int n0 = blockIdx.x * 64;
  const int tid = threadIdx.x, lane = tid & 31, wave = tid >> 5;
  const int l16 = lane & 15, h = (lane >> 4) & 1;
  const int mi = wave & 3, nj2 = wave >> 2;

  const bf16_t* xbb = xb + (size_t)b * C_CH * N_SP;

  v8f acc[2];
  acc[0] = v8f_zero();
  acc[1] = v8f_zero();

  for (int k0 = 0; k0 < N_SP; k0 += 32) {
    v16bf a = frag_a(xbb + (size_t)(m0 + mi * 16) * N_SP + k0, N_SP, lane);
#pragma unroll
    for (int t = 0; t < 2; ++t) {
      v16bf bf = frag_b(xbb + (size_t)(n0 + (nj2 * 2 + t) * 16) * N_SP + k0,
                        N_SP, lane);
      acc[t] = wmma_bf16(a, bf, acc[t]);
    }
  }

#pragma unroll
  for (int t = 0; t < 2; ++t)
#pragma unroll
    for (int r = 0; r < 8; ++r) {
      int c = m0 + mi * 16 + r + 8 * h;
      int d = n0 + (nj2 * 2 + t) * 16 + l16;
      camE[((size_t)b * C_CH + c) * C_CH + d] = acc[t][r];
    }
}

// ---------------------------------------------------------------------------
// K4: CAM softmax with max-trick: attn = softmax(rowmax(E) - E).
// One workgroup per row (512 cols, 2 per thread).
// ---------------------------------------------------------------------------
__global__ __launch_bounds__(256) void cam_softmax_kernel(
    const float* __restrict__ camE, bf16_t* __restrict__ camA) {
  const int b = blockIdx.y, c = blockIdx.x, tid = threadIdx.x;
  const float* er = camE + ((size_t)b * C_CH + c) * C_CH;
  __shared__ float red[256];

  float v0 = er[tid], v1 = er[tid + 256];

  red[tid] = fmaxf(v0, v1);
  __syncthreads();
  for (int s = 128; s > 0; s >>= 1) {
    if (tid < s) red[tid] = fmaxf(red[tid], red[tid + s]);
    __syncthreads();
  }
  float M = red[0];
  __syncthreads();

  float e0 = M - v0, e1 = M - v1;
  red[tid] = fmaxf(e0, e1);
  __syncthreads();
  for (int s = 128; s > 0; s >>= 1) {
    if (tid < s) red[tid] = fmaxf(red[tid], red[tid + s]);
    __syncthreads();
  }
  float M2 = red[0];
  __syncthreads();

  float p0 = __expf(e0 - M2), p1 = __expf(e1 - M2);
  red[tid] = p0 + p1;
  __syncthreads();
  for (int s = 128; s > 0; s >>= 1) {
    if (tid < s) red[tid] += red[tid + s];
    __syncthreads();
  }
  float inv = 1.0f / red[0];

  bf16_t* ar = camA + ((size_t)b * C_CH + c) * C_CH;
  ar[tid] = (bf16_t)(p0 * inv);
  ar[tid + 256] = (bf16_t)(p1 * inv);
}

// ---------------------------------------------------------------------------
// K5: CAM output. s12[c][m] += gamma_cam * (attn x)[c][m] + x[c][m]
// M=512, N=4096, K=512. A = camA[c][d] (global), B = xT[m][d] (global). LDS-free.
// ---------------------------------------------------------------------------
__global__ __launch_bounds__(256) void cam_out_kernel(
    const bf16_t* __restrict__ camA, const bf16_t* __restrict__ xbT,
    const float* __restrict__ x, const float* __restrict__ gamma_cam,
    float* __restrict__ s12) {
  const int b = blockIdx.z;
  const int m0 = blockIdx.y * 64;
  const int n0 = blockIdx.x * 128;
  const int tid = threadIdx.x, lane = tid & 31, wave = tid >> 5;
  const int l16 = lane & 15, h = (lane >> 4) & 1;
  const int mi = wave & 3, nh = wave >> 2;

  const bf16_t* xT = xbT + (size_t)b * N_SP * C_CH;
  const bf16_t* aw = camA + (size_t)b * C_CH * C_CH;

  v8f acc[4];
#pragma unroll
  for (int j = 0; j < 4; ++j) acc[j] = v8f_zero();

  for (int k0 = 0; k0 < C_CH; k0 += 32) {
    v16bf a = frag_a(aw + (size_t)(m0 + mi * 16) * C_CH + k0, C_CH, lane);
#pragma unroll
    for (int j = 0; j < 4; ++j) {
      v16bf bf = frag_b(xT + (size_t)(n0 + nh * 64 + j * 16) * C_CH + k0,
                        C_CH, lane);
      acc[j] = wmma_bf16(a, bf, acc[j]);
    }
  }

  const float g = gamma_cam[0];
#pragma unroll
  for (int j = 0; j < 4; ++j)
#pragma unroll
    for (int r = 0; r < 8; ++r) {
      int c = m0 + mi * 16 + r + 8 * h;
      int m = n0 + nh * 64 + j * 16 + l16;
      size_t idx = ((size_t)b * C_CH + c) * N_SP + m;
      s12[idx] += g * acc[j][r] + x[idx];
    }
}

// ---------------------------------------------------------------------------
// K6: final 1x1 conv. y[o][p] = conv1_w[o][:] . (x1+x2)[:][p]
// M=256, N=4096, K=512. s12 (f32) -> bf16 while staging LDS (float4 reads,
// fixed trip count). A = convw[o][c] direct from global.
// ---------------------------------------------------------------------------
__global__ __launch_bounds__(256) void conv_kernel(
    const bf16_t* __restrict__ convw, const float* __restrict__ s12,
    float* __restrict__ y) {
  const int b = blockIdx.z;
  const int m0 = blockIdx.y * 64;
  const int n0 = blockIdx.x * 128;
  const int tid = threadIdx.x, lane = tid & 31, wave = tid >> 5;
  const int l16 = lane & 15, h = (lane >> 4) & 1;
  const int mi = wave & 3, nh = wave >> 2;

  __shared__ bf16_t St[128 * 32];  // [p][c] transposed tile

  const float* sb = s12 + (size_t)b * C_CH * N_SP;

  v8f acc[4];
#pragma unroll
  for (int j = 0; j < 4; ++j) acc[j] = v8f_zero();

  for (int k0 = 0; k0 < C_CH; k0 += 32) {
    __syncthreads();
#pragma unroll
    for (int it = 0; it < 4; ++it) {
      int i = tid + it * 256;            // 0..1023
      int kk = i >> 5, n4 = (i & 31) * 4;
      float4 v = *(const float4*)(sb + (size_t)(k0 + kk) * N_SP + n0 + n4);
      St[(n4 + 0) * 32 + kk] = (bf16_t)v.x;
      St[(n4 + 1) * 32 + kk] = (bf16_t)v.y;
      St[(n4 + 2) * 32 + kk] = (bf16_t)v.z;
      St[(n4 + 3) * 32 + kk] = (bf16_t)v.w;
    }
    __syncthreads();
    v16bf a = frag_a(convw + (size_t)(m0 + mi * 16) * C_CH + k0, C_CH, lane);
#pragma unroll
    for (int j = 0; j < 4; ++j) {
      v16bf bf = frag_b(St + (nh * 64 + j * 16) * 32, 32, lane);
      acc[j] = wmma_bf16(a, bf, acc[j]);
    }
  }

#pragma unroll
  for (int j = 0; j < 4; ++j)
#pragma unroll
    for (int r = 0; r < 8; ++r) {
      int o = m0 + mi * 16 + r + 8 * h;
      int p = n0 + nh * 64 + j * 16 + l16;
      y[((size_t)b * OUTC + o) * N_SP + p] = acc[j][r];
    }
}

// ---------------------------------------------------------------------------
// K7: BN batch stats. One workgroup per channel.
// ---------------------------------------------------------------------------
__global__ __launch_bounds__(256) void bn_stats_kernel(
    const float* __restrict__ y, float* __restrict__ stats) {
  const int o = blockIdx.x, tid = threadIdx.x;
  __shared__ float rs[256], rs2[256];
  float s = 0.0f, s2 = 0.0f;
  for (int b = 0; b < NB; ++b) {
    const float* yp = y + ((size_t)b * OUTC + o) * N_SP;
#pragma unroll
    for (int it = 0; it < 4; ++it) {
      float4 v = *(const float4*)(yp + (tid + it * 256) * 4);
      s += v.x + v.y + v.z + v.w;
      s2 += v.x * v.x + v.y * v.y + v.z * v.z + v.w * v.w;
    }
  }
  rs[tid] = s;
  rs2[tid] = s2;
  __syncthreads();
  for (int st = 128; st > 0; st >>= 1) {
    if (tid < st) { rs[tid] += rs[tid + st]; rs2[tid] += rs2[tid + st]; }
    __syncthreads();
  }
  if (tid == 0) {
    const float inv_n = 1.0f / (float)(NB * N_SP);
    float mean = rs[0] * inv_n;
    float var = rs2[0] * inv_n - mean * mean;
    stats[o] = mean;
    stats[OUTC + o] = rsqrtf(var + 1e-5f);
  }
}

// ---------------------------------------------------------------------------
// K8: BN apply + ReLU -> d_out
// ---------------------------------------------------------------------------
__global__ __launch_bounds__(256) void bn_apply_kernel(
    const float* __restrict__ y, const float* __restrict__ stats,
    const float* __restrict__ bn_gamma, const float* __restrict__ bn_beta,
    float* __restrict__ out) {
  size_t idx = (size_t)blockIdx.x * 256 + threadIdx.x;
  int o = (int)((idx >> 12) & (OUTC - 1));  // layout [b][o][p], p = 4096
  float v = (y[idx] - stats[o]) * stats[OUTC + o] * bn_gamma[o] + bn_beta[o];
  out[idx] = fmaxf(v, 0.0f);
}

// ---------------------------------------------------------------------------
// Launch
// ---------------------------------------------------------------------------
extern "C" void kernel_launch(void* const* d_in, const int* in_sizes, int n_in,
                              void* d_out, int out_size, void* d_ws, size_t ws_size,
                              hipStream_t stream) {
  const float* x         = (const float*)d_in[0];
  const float* q_w       = (const float*)d_in[1];
  const float* q_b       = (const float*)d_in[2];
  const float* k_w       = (const float*)d_in[3];
  const float* k_b       = (const float*)d_in[4];
  const float* v_w       = (const float*)d_in[5];
  const float* v_b       = (const float*)d_in[6];
  const float* gamma_pam = (const float*)d_in[7];
  const float* gamma_cam = (const float*)d_in[8];
  const float* conv1_w   = (const float*)d_in[9];
  const float* bn_gamma  = (const float*)d_in[10];
  const float* bn_beta   = (const float*)d_in[11];
  float* out = (float*)d_out;

  char* ws = (char*)d_ws;
  // workspace layout (bytes, all 256-aligned); total ~112 MB
  constexpr size_t OFF_XB    = 0;                       // [B][C][N] bf16, 16 MB
  constexpr size_t OFF_XBT   = OFF_XB    + 16777216;    // [B][N][C] bf16, 16 MB
  constexpr size_t OFF_QKT   = OFF_XBT   + 16777216;    // [B][2][N][64] bf16, 4 MB
  constexpr size_t OFF_VB    = OFF_QKT   + 4194304;     // [B][C][N] bf16, 16 MB
  constexpr size_t OFF_WQKV  = OFF_VB    + 16777216;    // 640*512 bf16
  constexpr size_t OFF_BIAS  = OFF_WQKV  + 655360;      // 640 f32
  constexpr size_t OFF_CONVW = OFF_BIAS  + 2560;        // 256*512 bf16
  constexpr size_t OFF_S12   = OFF_CONVW + 262144;      // [B][C][N] f32, 32 MB
  constexpr size_t OFF_CAME  = OFF_S12   + 33554432;    // [B][512][512] f32
  constexpr size_t OFF_CAMA  = OFF_CAME  + 4194304;     // [B][512][512] bf16
  constexpr size_t OFF_Y     = OFF_CAMA  + 2097152;     // [B][256][N] f32, 16 MB
  constexpr size_t OFF_BN    = OFF_Y     + 16777216;    // 512 f32

  bf16_t* xb      = (bf16_t*)(ws + OFF_XB);
  bf16_t* xbT     = (bf16_t*)(ws + OFF_XBT);
  bf16_t* qkT     = (bf16_t*)(ws + OFF_QKT);
  bf16_t* vB      = (bf16_t*)(ws + OFF_VB);
  bf16_t* wqkv    = (bf16_t*)(ws + OFF_WQKV);
  float*  bias640 = (float*)(ws + OFF_BIAS);
  bf16_t* convw   = (bf16_t*)(ws + OFF_CONVW);
  float*  s12     = (float*)(ws + OFF_S12);
  float*  camE    = (float*)(ws + OFF_CAME);
  bf16_t* camA    = (bf16_t*)(ws + OFF_CAMA);
  float*  y       = (float*)(ws + OFF_Y);
  float*  bnstat  = (float*)(ws + OFF_BN);

  // K0: casts / transpose / weight packing
  cast_transpose_kernel<<<dim3(N_SP / 32, C_CH / 32, NB), dim3(32, 8), 0,
                          stream>>>(x, xb, xbT);
  pack_w_kernel<<<(640 * 512 + 255) / 256, 256, 0, stream>>>(
      q_w, k_w, v_w, q_b, k_b, v_b, wqkv, bias640);
  cast_f32_bf16_kernel<<<(OUTC * C_CH) / 256, 256, 0, stream>>>(
      conv1_w, convw, OUTC * C_CH);

  // K1: QKV projection (writes q/k transposed, v row-major)
  qkv_kernel<<<dim3(N_SP / 128, 640 / 64, NB), 256, 0, stream>>>(
      wqkv, xbT, bias640, qkT, vB);

  // K2: fused PAM (flash attention)
  pam_kernel<<<dim3(N_SP / 32, NB), 256, 0, stream>>>(qkT, vB, x, gamma_pam,
                                                      s12);

  // K3-K5: CAM
  cam_energy_kernel<<<dim3(C_CH / 64, C_CH / 64, NB), 256, 0, stream>>>(xb,
                                                                        camE);
  cam_softmax_kernel<<<dim3(C_CH, NB), 256, 0, stream>>>(camE, camA);
  cam_out_kernel<<<dim3(N_SP / 128, C_CH / 64, NB), 256, 0, stream>>>(
      camA, xbT, x, gamma_cam, s12);

  // K6: final 1x1 conv
  conv_kernel<<<dim3(N_SP / 128, OUTC / 64, NB), 256, 0, stream>>>(convw, s12,
                                                                   y);

  // K7-K8: batchnorm (training-mode batch stats) + ReLU
  bn_stats_kernel<<<OUTC, 256, 0, stream>>>(y, bnstat);
  bn_apply_kernel<<<(NB * OUTC * N_SP) / 256, 256, 0, stream>>>(
      y, bnstat, bn_gamma, bn_beta, out);
}
